// MYLSTM_76046690943028
// MI455X (gfx1250) — matrix operand order repaired and checked by reading, not compiled, not run on hardware
//
#include <hip/hip_runtime.h>
#include <hip/hip_bf16.h>
#include <math.h>

// ---------------------------------------------------------------------------
// Social-LSTM trajectory predictor for MI455X (gfx1250, wave32).
//
// Dominant cost: pooling MLP over N*N pairs, 12 decoder steps.
// Factorized:  z[i,j] = relu(R[i] + S[j])  (64-dim),  pool = relu(z @ W2^T + b2)
// The 64->8 pairwise GEMM runs on v_wmma_f32_16x16x32_f16 (two K-steps),
// A-fragments built with v_pk_add_f16 / v_pk_max in WMMA register layout,
// S[j] broadcast across the wave with ds_bpermute (__shfl), masked max over j
// done as bitwise u32 atomicMax (valid: all pooled values are >= 0).
// ---------------------------------------------------------------------------

#define NN 1024
#define HH 8
#define T_PRED 12
#define OBS_LEN 8

typedef _Float16 v16h __attribute__((ext_vector_type(16)));
typedef float    v8f  __attribute__((ext_vector_type(8)));
typedef _Float16 h2   __attribute__((ext_vector_type(2)));

// workspace layout (units: 32-bit words from start of d_ws)
#define WS_H       0        // [N*8] lstm hidden
#define WS_C       8192     // [N*8] lstm cell
#define WS_POOLH   16384    // [N*8] pooled context (atomic-max target)
#define WS_POSPREV 24576    // [N*2] previous position
#define WS_WC0     26624    // [64]  folded Wsp->W1, pos-x
#define WS_WC1     26688    // [64]  folded Wsp->W1, pos-y
#define WS_SBIAS   26752    // [64]  bsp@W1[:, :32]^T (b1 added separately)
#define WS_RH_U32  26816    // [N*32] u32 = f16x2 packed R rows
#define WS_SH_U32  59584    // [N*32] u32 = f16x2 packed S rows

__device__ __forceinline__ float sigf(float x) { return 1.0f / (1.0f + expf(-x)); }

// ---------------------------------------------------------------------------
// One-time: fold spatial embedding into W1 (rank-2 position term + bias term)
// ---------------------------------------------------------------------------
__global__ void k_init(const float* __restrict__ Wsp, const float* __restrict__ bsp,
                       const float* __restrict__ W1, float* __restrict__ ws) {
  int e = threadIdx.x;
  if (e < 64) {
    float wc0 = 0.f, wc1 = 0.f, sb = 0.f;
    for (int s = 0; s < 32; ++s) {
      float w = W1[e * 48 + s];
      wc0 += Wsp[s * 2 + 0] * w;
      wc1 += Wsp[s * 2 + 1] * w;
      sb  += bsp[s] * w;
    }
    ws[WS_WC0 + e] = wc0;
    ws[WS_WC1 + e] = wc1;
    ws[WS_SBIAS + e] = sb;
  }
}

// ---------------------------------------------------------------------------
// Encoder: 8 LSTM steps per agent; also zero-initialize decoder carry state.
// ---------------------------------------------------------------------------
__global__ void k_encoder(const float* __restrict__ obs_traj,
                          const float* __restrict__ h0, const float* __restrict__ c0,
                          const float* __restrict__ We, const float* __restrict__ be,
                          const float* __restrict__ Wih, const float* __restrict__ Whh,
                          const float* __restrict__ bih, const float* __restrict__ bhh,
                          float* __restrict__ ws) {
  int n = blockIdx.x * blockDim.x + threadIdx.x;
  if (n >= NN) return;
  float h[HH], c[HH];
  for (int q = 0; q < HH; ++q) { h[q] = h0[n * HH + q]; c[q] = c0[n * HH + q]; }
  for (int t = 0; t < OBS_LEN; ++t) {
    float px = obs_traj[((size_t)t * NN + n) * 2 + 0];
    float py = obs_traj[((size_t)t * NN + n) * 2 + 1];
    float x[16];
    for (int k = 0; k < 16; ++k)
      x[k] = fmaxf(We[k * 2 + 0] * px + We[k * 2 + 1] * py + be[k], 0.f);
    float g[32];
    for (int r = 0; r < 32; ++r) {
      float a = bih[r] + bhh[r];
      for (int k = 0; k < 16; ++k) a += Wih[r * 16 + k] * x[k];
      for (int q = 0; q < HH; ++q) a += Whh[r * HH + q] * h[q];
      g[r] = a;
    }
    for (int q = 0; q < HH; ++q) {
      float ig = sigf(g[q]), fg = sigf(g[8 + q]);
      float gg = tanhf(g[16 + q]), og = sigf(g[24 + q]);
      c[q] = fg * c[q] + ig * gg;
      h[q] = og * tanhf(c[q]);
    }
  }
  for (int q = 0; q < HH; ++q) {
    ws[WS_H + n * HH + q] = h[q];
    ws[WS_C + n * HH + q] = 0.f;       // decoder starts with c = 0
    ws[WS_POOLH + n * HH + q] = 0.f;   // context0 = 0
  }
  ws[WS_POSPREV + n * 2 + 0] = 0.f;    // output0 = 0
  ws[WS_POSPREV + n * 2 + 1] = 0.f;
}

// ---------------------------------------------------------------------------
// Per decoder step, phase 1: decoder input MLP + LSTM cell, then build the
// separable pooling rows R[i], S[j] (f16x2-packed for WMMA A-fragments).
// Each thread owns agent n; it also re-zeroes its pool_h slots (after
// consuming them as context) so the pooling atomic-max starts from 0.
// ---------------------------------------------------------------------------
__global__ void k_step_pre(int t, const float* __restrict__ obs_obs,
                           const float* __restrict__ Wd, const float* __restrict__ bd,
                           const float* __restrict__ Wih, const float* __restrict__ Whh,
                           const float* __restrict__ bih, const float* __restrict__ bhh,
                           const float* __restrict__ W1, const float* __restrict__ b1,
                           float* __restrict__ ws) {
  int n = blockIdx.x * blockDim.x + threadIdx.x;
  if (n >= NN) return;
  (void)t;

  float cat[10];
  for (int q = 0; q < 8; ++q) {
    cat[q] = ws[WS_POOLH + n * 8 + q];
    ws[WS_POOLH + n * 8 + q] = 0.f;    // reset atomic-max accumulator
  }
  cat[8] = ws[WS_POSPREV + n * 2 + 0];
  cat[9] = ws[WS_POSPREV + n * 2 + 1];

  float x[16];
  for (int k = 0; k < 16; ++k) {
    float a = bd[k];
    for (int j = 0; j < 10; ++j) a += Wd[k * 10 + j] * cat[j];
    x[k] = fmaxf(a, 0.f);
  }

  float h[HH], c[HH];
  for (int q = 0; q < HH; ++q) { h[q] = ws[WS_H + n * HH + q]; c[q] = ws[WS_C + n * HH + q]; }
  float g[32];
  for (int r = 0; r < 32; ++r) {
    float a = bih[r] + bhh[r];
    for (int k = 0; k < 16; ++k) a += Wih[r * 16 + k] * x[k];
    for (int q = 0; q < HH; ++q) a += Whh[r * HH + q] * h[q];
    g[r] = a;
  }
  for (int q = 0; q < HH; ++q) {
    float ig = sigf(g[q]), fg = sigf(g[8 + q]);
    float gg = tanhf(g[16 + q]), og = sigf(g[24 + q]);
    c[q] = fg * c[q] + ig * gg;
    h[q] = og * tanhf(c[q]);
    ws[WS_H + n * HH + q] = h[q];
    ws[WS_C + n * HH + q] = c[q];
  }

  // R[n][e] =  pos_n . Wc[:,e] + h_n . W1[e, 40:48]   (the "i" half)
  // S[n][e] = -pos_n . Wc[:,e] + h_n . W1[e, 32:40] + b1[e] + SBIAS[e]  ("j" half)
  float p0 = obs_obs[((size_t)(OBS_LEN - 1) * NN + n) * 2 + 0];
  float p1 = obs_obs[((size_t)(OBS_LEN - 1) * NN + n) * 2 + 1];
  unsigned int* Rh = (unsigned int*)ws + WS_RH_U32;
  unsigned int* Sh = (unsigned int*)ws + WS_SH_U32;
  for (int p = 0; p < 32; ++p) {
    h2 rp, sp;
    for (int half = 0; half < 2; ++half) {
      int e = 2 * p + half;
      float pdot = p0 * ws[WS_WC0 + e] + p1 * ws[WS_WC1 + e];
      float r = pdot;
      float s = -pdot + b1[e] + ws[WS_SBIAS + e];
      for (int q = 0; q < HH; ++q) {
        r += h[q] * W1[e * 48 + 40 + q];
        s += h[q] * W1[e * 48 + 32 + q];
      }
      rp[half] = (_Float16)r;
      sp[half] = (_Float16)s;
    }
    Rh[n * 32 + p] = __builtin_bit_cast(unsigned int, rp);
    Sh[n * 32 + p] = __builtin_bit_cast(unsigned int, sp);
  }
}

// ---------------------------------------------------------------------------
// Per decoder step, phase 2: pairwise pooling with WMMA.
// Block = 16-row i-tile x 128-column j-segment, 8 waves, 1 j per wave per iter.
// A (16x32 f16, per ISA layout): lanes 0-15 m=lane  k={2v,2v+1}+{0|8}+hi*8
// B (32x16 f16, same k order as A): lanes 0-15 k=0-7,16-23; lanes 16-31 k=8-15,24-31
// C/D (16x16 f32): lane -> col n = lane&15, VGPR r -> row m = r + 8*(lane>>4)
// ---------------------------------------------------------------------------
__global__ void __launch_bounds__(256) k_pool(int t, const int* __restrict__ nei,
                                              const float* __restrict__ W2,
                                              const float* __restrict__ b2,
                                              float* __restrict__ ws) {
  const unsigned int* Rh = (const unsigned int*)ws + WS_RH_U32;
  const unsigned int* Sh = (const unsigned int*)ws + WS_SH_U32;
  unsigned int* poolhU = (unsigned int*)(ws + WS_POOLH);

  const int lane = threadIdx.x & 31;
  const int wave = threadIdx.x >> 5;
  const int nn   = lane & 15;   // D column; also A row index m
  const int hi   = lane >> 4;
  const int iBase = blockIdx.x * 16;

  // --- constant B fragments (W2 zero-padded 8 -> 16 output columns) ---
  v16h bf0, bf1;
  for (int v = 0; v < 8; ++v)
    for (int half = 0; half < 2; ++half) {
      int kl = 2 * v + half + ((v >= 4) ? 8 : 0) + hi * 8;
      float w0 = (nn < 8) ? W2[nn * 64 + kl]      : 0.f;
      float w1 = (nn < 8) ? W2[nn * 64 + 32 + kl] : 0.f;
      bf0[2 * v + half] = (_Float16)w0;
      bf1[2 * v + half] = (_Float16)w1;
    }

  // --- per-lane f16x2 pack indices for the A fragment (k = 2*pb, 2*pb+1) ---
  int pb[8];
  for (int v = 0; v < 8; ++v) pb[v] = v + ((v >= 4) ? 4 : 0) + hi * 4;

  // --- j-invariant R packs, hoisted out of the loop ---
  unsigned int rpk[16];
  for (int v = 0; v < 8; ++v) {
    rpk[v]     = Rh[(iBase + nn) * 32 + pb[v]];
    rpk[8 + v] = Rh[(iBase + nn) * 32 + pb[v] + 16];
  }

  const float b2n = (nn < 8) ? b2[nn] : 0.f;
  float acc[8];
  for (int r = 0; r < 8; ++r) acc[r] = 0.f;

  const size_t tOff = (size_t)t * NN * NN;
  const int j0 = blockIdx.y * 128 + wave;   // 8 waves stride the 128-j segment

  for (int it = 0; it < 16; ++it) {
    const int j = j0 + it * 8;
    const unsigned int spMine = Sh[j * 32 + lane];   // lane L holds S pack L
    if (it + 1 < 16) __builtin_prefetch(&Sh[(j + 8) * 32 + lane], 0, 1);

    // A = relu(R + S), assembled directly in WMMA layout (packed f16 math)
    v16h a0, a1;
    const h2 zero = {};
    for (int v = 0; v < 8; ++v) {
      unsigned int s0 = (unsigned int)__shfl((int)spMine, pb[v], 32);
      unsigned int s1 = (unsigned int)__shfl((int)spMine, pb[v] + 16, 32);
      h2 z0 = __builtin_bit_cast(h2, rpk[v])     + __builtin_bit_cast(h2, s0);
      h2 z1 = __builtin_bit_cast(h2, rpk[8 + v]) + __builtin_bit_cast(h2, s1);
      z0 = __builtin_elementwise_max(z0, zero);
      z1 = __builtin_elementwise_max(z1, zero);
      a0[2 * v] = z0[0]; a0[2 * v + 1] = z0[1];
      a1[2 * v] = z1[0]; a1[2 * v + 1] = z1[1];
    }

    v8f d = {0.f, 0.f, 0.f, 0.f, 0.f, 0.f, 0.f, 0.f};
    d = __builtin_amdgcn_wmma_f32_16x16x32_f16(false, a0, false, bf0, (short)0, d, false, false);
    d = __builtin_amdgcn_wmma_f32_16x16x32_f16(false, a1, false, bf1, (short)0, d, false, false);

    // neighbor mask per tile row via ballot (bit m == mask for row m)
    int neiv = nei[tOff + (size_t)(iBase + nn) * NN + j];
    unsigned int bal = (unsigned int)__ballot(neiv > 0);

    for (int r = 0; r < 8; ++r) {
      float pv = fmaxf(d[r] + b2n, 0.f);                 // bias + relu
      unsigned int mk = (bal >> (r + 8 * hi)) & 1u;      // nei[i, j] > 0 ?
      acc[r] = fmaxf(acc[r], mk ? pv : 0.f);             // masked running max
    }
  }

  // all pooled values >= 0 -> float max == u32 max on bit patterns
  if (nn < 8) {
    for (int r = 0; r < 8; ++r)
      atomicMax(&poolhU[(size_t)(iBase + r + 8 * hi) * 8 + nn],
                __float_as_uint(acc[r]));
  }
}

// ---------------------------------------------------------------------------
// Per decoder step, phase 3: output heads + reparameterized position.
// ---------------------------------------------------------------------------
__global__ void k_step_post(int t, const float* __restrict__ eps,
                            const float* __restrict__ Wm, const float* __restrict__ bm,
                            const float* __restrict__ Wv, const float* __restrict__ bv,
                            float* __restrict__ ws, float* __restrict__ out) {
  int n = blockIdx.x * blockDim.x + threadIdx.x;
  if (n >= NN) return;
  float pm[12], pl[12];
  for (int q = 0; q < 4; ++q) {
    pm[q] = ws[WS_H + n * 8 + q];
    pl[q] = ws[WS_H + n * 8 + 4 + q];
  }
  for (int q = 0; q < 8; ++q) {
    float ph = ws[WS_POOLH + n * 8 + q];
    pm[4 + q] = ph;
    pl[4 + q] = ph;
  }
  const size_t base = ((size_t)t * NN + n) * 2;
  for (int d = 0; d < 2; ++d) {
    float mu = bm[d], lv = bv[d];
    for (int k = 0; k < 12; ++k) {
      mu += Wm[d * 12 + k] * pm[k];
      lv += Wv[d * 12 + k] * pl[k];
    }
    float e = eps[base + d];
    float pos = mu + e * expf(0.5f * lv);
    out[base + d] = pos;                                  // pred_traj
    out[(size_t)T_PRED * NN * 2 + base + d] = mu;         // mean
    out[(size_t)2 * T_PRED * NN * 2 + base + d] = lv;     // var (logvar)
    ws[WS_POSPREV + n * 2 + d] = pos;
  }
}

// ---------------------------------------------------------------------------
extern "C" void kernel_launch(void* const* d_in, const int* in_sizes, int n_in,
                              void* d_out, int out_size, void* d_ws, size_t ws_size,
                              hipStream_t stream) {
  (void)in_sizes; (void)n_in; (void)out_size; (void)ws_size;
  const float* obs_traj = (const float*)d_in[0];
  const float* obs_obs  = (const float*)d_in[1];
  const int*   nei      = (const int*)d_in[2];
  // d_in[3] nei_num_index: unused by the reference body
  const float* h0  = (const float*)d_in[4];
  const float* c0  = (const float*)d_in[5];
  const float* eps = (const float*)d_in[6];
  const float* We = (const float*)d_in[7];   const float* be = (const float*)d_in[8];
  const float* Wih_t = (const float*)d_in[9];  const float* Whh_t = (const float*)d_in[10];
  const float* bih_t = (const float*)d_in[11]; const float* bhh_t = (const float*)d_in[12];
  const float* Wd = (const float*)d_in[13];  const float* bd = (const float*)d_in[14];
  const float* Wih_p = (const float*)d_in[15]; const float* Whh_p = (const float*)d_in[16];
  const float* bih_p = (const float*)d_in[17]; const float* bhh_p = (const float*)d_in[18];
  const float* Wsp = (const float*)d_in[19]; const float* bsp = (const float*)d_in[20];
  const float* W1 = (const float*)d_in[21];  const float* b1 = (const float*)d_in[22];
  const float* W2 = (const float*)d_in[23];  const float* b2 = (const float*)d_in[24];
  const float* Wm = (const float*)d_in[25];  const float* bm = (const float*)d_in[26];
  const float* Wv = (const float*)d_in[27];  const float* bv = (const float*)d_in[28];

  float* ws  = (float*)d_ws;
  float* out = (float*)d_out;

  k_init<<<1, 64, 0, stream>>>(Wsp, bsp, W1, ws);
  k_encoder<<<4, 256, 0, stream>>>(obs_traj, h0, c0, We, be,
                                   Wih_t, Whh_t, bih_t, bhh_t, ws);
  for (int t = 0; t < T_PRED; ++t) {
    k_step_pre<<<4, 256, 0, stream>>>(t, obs_obs, Wd, bd,
                                      Wih_p, Whh_p, bih_p, bhh_p, W1, b1, ws);
    k_pool<<<dim3(NN / 16, 8), 256, 0, stream>>>(t, nei, W2, b2, ws);
    k_step_post<<<4, 256, 0, stream>>>(t, eps, Wm, bm, Wv, bv, ws, out);
  }
}
